// split_loss_74440373174914
// MI455X (gfx1250) — matrix operand order repaired and checked by reading, not compiled, not run on hardware
//
#include <hip/hip_runtime.h>
#include <hip/hip_bf16.h>
#include <math.h>

typedef __attribute__((ext_vector_type(16))) __bf16          v16bf;
typedef __attribute__((ext_vector_type(8)))  float           v8f;
typedef __attribute__((ext_vector_type(8)))  unsigned short  v8us;
typedef __attribute__((ext_vector_type(4)))  float           v4f;
typedef __attribute__((ext_vector_type(4)))  unsigned short  v4us;
typedef __attribute__((ext_vector_type(4)))  int             v4i;

// address-space-qualified int4 pointers for the async-copy builtin
typedef __attribute__((address_space(1))) v4i gas_v4i;
typedef __attribute__((address_space(3))) v4i las_v4i;

#define DK     512
#define NROWS  4096
#define MT     32
#define NT     256
#define KC     64    // K elements staged per barrier round (2 WMMA k-steps)
#define NTILES 199   // 20 (head) + 40 + 59 + 80 (tails)

#if defined(__gfx1250__) && __has_builtin(__builtin_amdgcn_global_load_async_to_lds_b128)
#define ASYNC_COPY 1
#else
#define ASYNC_COPY 0
#endif

__device__ __forceinline__ unsigned short f2bf(float f) {
  unsigned u = __float_as_uint(f);
  u += 0x7FFFu + ((u >> 16) & 1u);   // round-to-nearest-even
  return (unsigned short)(u >> 16);
}

__device__ __forceinline__ void copy16_g2l(const unsigned short* g, unsigned short* l) {
#if ASYNC_COPY
  __builtin_amdgcn_global_load_async_to_lds_b128((gas_v4i*)g, (las_v4i*)l, 0, 0);
#else
  *(v8us*)l = *(const v8us*)g;
#endif
}

__device__ __forceinline__ void async_copy_wait() {
#if ASYNC_COPY
#if __has_builtin(__builtin_amdgcn_s_wait_asynccnt)
  __builtin_amdgcn_s_wait_asynccnt(0);
#else
  asm volatile("s_wait_asynccnt 0x0" ::: "memory");
#endif
#endif
}

union Frag { v8us u[2]; v16bf b; };

// ---------------------------------------------------------------------------
// Phase 0: one-shot f32 -> bf16 conversion (weight + hidden) into workspace.
// ---------------------------------------------------------------------------
__global__ __launch_bounds__(256) void cvt_bf16_kernel(
    const float* __restrict__ src, unsigned short* __restrict__ dst, int n4)
{
  int i = blockIdx.x * 256 + threadIdx.x;
  if (i < n4) {
    v4f v = *(const v4f*)&src[(size_t)i * 4];
    v4us o = { f2bf(v.x), f2bf(v.y), f2bf(v.z), f2bf(v.w) };
    *(v4us*)&dst[(size_t)i * 4] = o;
  }
}

// ---------------------------------------------------------------------------
// Phase 1: bf16 WMMA GEMM (logits = hidden @ W^T + bias) fused with
// per-(tile,row) max / sum-exp partials; async global->LDS staging.
// ---------------------------------------------------------------------------
__global__ __launch_bounds__(256) void adasoft_gemm_lse(
    const unsigned short* __restrict__ hb, const unsigned short* __restrict__ wb,
    const float* __restrict__ bias, float* __restrict__ part)
{
  __shared__ __align__(16) unsigned short sA[MT * KC];   // [row][k] bf16, 4 KB
  __shared__ __align__(16) unsigned short sB[NT * KC];   // [col][k] bf16, 32 KB
  __shared__ float sRed[8][MT][2];                       // per-wave (max,sum)

  const int tid      = threadIdx.x;
  const int wave     = tid >> 5;
  const int lane     = tid & 31;
  const int lane16   = lane & 15;
  const int laneHalf = lane >> 4;

  // map blockIdx.x -> (segment, tile); tiles never cross a softmax boundary
  const int bx = blockIdx.x;
  int seg, t;
  if      (bx < 20)  { seg = 0; t = bx; }
  else if (bx < 60)  { seg = 1; t = bx - 20; }
  else if (bx < 119) { seg = 2; t = bx - 60; }
  else               { seg = 3; t = bx - 119; }
  const int segStart[4] = {0, 5000, 15000, 30000};
  const int segEnd[4]   = {5000, 15000, 30000, 50257};
  const int colBase = segStart[seg] + t * NT;
  const int count   = min(NT, segEnd[seg] - colBase);
  const int rowBase = blockIdx.y * MT;

  // staging decomposition: 256 threads x 16B, 1 chunk for A, 8 for B
  const int trow = tid >> 3;          // 0..31
  const int tk8  = (tid & 7) * 8;     // 0..56 step 8 (bf16 elements)

  v8f acc[2][2] = {};

  for (int k0 = 0; k0 < DK; k0 += KC) {
    __syncthreads();
    // A tile: 32 rows x 64 k
    copy16_g2l(&hb[(size_t)(rowBase + trow) * DK + k0 + tk8], &sA[trow * KC + tk8]);
    // B tile: 256 cols x 64 k; clamp OOB cols to a valid row (masked later)
#pragma unroll
    for (int i = 0; i < 8; ++i) {
      int c  = i * 32 + trow;
      int cc = min(c, count - 1);
      copy16_g2l(&wb[(size_t)(colBase + cc) * DK + k0 + tk8], &sB[c * KC + tk8]);
    }
    async_copy_wait();
    __syncthreads();

#pragma unroll
    for (int ks = 0; ks < 2; ++ks) {
      const int ko = ks * 32;
      Frag a[2], b[2];
#pragma unroll
      for (int mi = 0; mi < 2; ++mi) {
        int row = mi * 16 + lane16;
        int off = ko + laneHalf * 8;             // lanes 16-31: K 8..15,24..31
        a[mi].u[0] = *(const v8us*)&sA[row * KC + off];
        a[mi].u[1] = *(const v8us*)&sA[row * KC + 16 + off];
      }
#pragma unroll
      for (int ni = 0; ni < 2; ++ni) {
        int col = wave * 32 + ni * 16 + lane16;
        int off = ko + laneHalf * 8;             // lanes 16-31: K 8..15,24..31
        b[ni].u[0] = *(const v8us*)&sB[col * KC + off];
        b[ni].u[1] = *(const v8us*)&sB[col * KC + off + 16];
      }
      acc[0][0] = __builtin_amdgcn_wmma_f32_16x16x32_bf16(false, a[0].b, false, b[0].b, (short)0, acc[0][0], false, false);
      acc[0][1] = __builtin_amdgcn_wmma_f32_16x16x32_bf16(false, a[0].b, false, b[1].b, (short)0, acc[0][1], false, false);
      acc[1][0] = __builtin_amdgcn_wmma_f32_16x16x32_bf16(false, a[1].b, false, b[0].b, (short)0, acc[1][0], false, false);
      acc[1][1] = __builtin_amdgcn_wmma_f32_16x16x32_bf16(false, a[1].b, false, b[1].b, (short)0, acc[1][1], false, false);
    }
  }

  // ---- epilogue: bias + mask + per-row max/sumexp over this wave's 32 cols ----
  const int  c0 = wave * 32 + lane16;
  const int  c1 = c0 + 16;
  const bool v0 = c0 < count;
  const bool v1 = c1 < count;
  const float b0 = v0 ? bias[colBase + c0] : 0.0f;
  const float b1 = v1 ? bias[colBase + c1] : 0.0f;

#pragma unroll
  for (int mi = 0; mi < 2; ++mi) {
    float fm[8], fs[8];
#pragma unroll
    for (int j = 0; j < 8; ++j) {
      float x0 = v0 ? (acc[mi][0][j] + b0) : -INFINITY;
      float x1 = v1 ? (acc[mi][1][j] + b1) : -INFINITY;
      float m = fmaxf(x0, x1);
#pragma unroll
      for (int d = 1; d < 16; d <<= 1) m = fmaxf(m, __shfl_xor(m, d, 16));
      float s = 0.0f;
      if (x0 > -INFINITY) s += __expf(x0 - m);
      if (x1 > -INFINITY) s += __expf(x1 - m);
#pragma unroll
      for (int d = 1; d < 16; d <<= 1) s += __shfl_xor(s, d, 16);
      fm[j] = m; fs[j] = s;
    }
    if (lane16 == 0) {
      int rb = mi * 16 + laneHalf * 8;           // rows per C-matrix layout
#pragma unroll
      for (int j = 0; j < 8; ++j) {
        sRed[wave][rb + j][0] = fm[j];
        sRed[wave][rb + j][1] = fs[j];
      }
    }
  }
  __syncthreads();

  if (tid < MT) {
    int row = tid;
    float m = -INFINITY;
#pragma unroll
    for (int w = 0; w < 8; ++w) m = fmaxf(m, sRed[w][row][0]);
    float s = 0.0f;
#pragma unroll
    for (int w = 0; w < 8; ++w) {
      float wm = sRed[w][row][0];
      if (wm > -INFINITY) s += sRed[w][row][1] * __expf(wm - m);
    }
    size_t idx = ((size_t)bx * NROWS + rowBase + row) * 2;
    part[idx]     = m;
    part[idx + 1] = s;
  }
}

// ---------------------------------------------------------------------------
// Phase 2: one wave per row. Combine partials into 4 segment LSEs, recompute
// routing logits + target logit (f32 dots), assemble NLL.
// ---------------------------------------------------------------------------
__global__ __launch_bounds__(256) void adasoft_finalize(
    const float* __restrict__ hidden, const int* __restrict__ target,
    const float* __restrict__ weight, const float* __restrict__ bias,
    const float* __restrict__ cw, const float* __restrict__ cbias,
    const float* __restrict__ part, float* __restrict__ out)
{
  const int wave = threadIdx.x >> 5;
  const int lane = threadIdx.x & 31;
  const int row  = blockIdx.x * 8 + wave;
  if (row >= NROWS) return;

  const int segOff[4] = {0, 20, 60, 119};
  const int segCnt[4] = {20, 40, 59, 80};
  float lm[4], ls[4];
#pragma unroll
  for (int s = 0; s < 4; ++s) {
    float m = -INFINITY;
    for (int tt = lane; tt < segCnt[s]; tt += 32)
      m = fmaxf(m, part[((size_t)(segOff[s] + tt) * NROWS + row) * 2]);
#pragma unroll
    for (int d = 1; d < 32; d <<= 1) m = fmaxf(m, __shfl_xor(m, d, 32));
    float sum = 0.0f;
    for (int tt = lane; tt < segCnt[s]; tt += 32) {
      size_t idx = ((size_t)(segOff[s] + tt) * NROWS + row) * 2;
      float tm = part[idx], tsv = part[idx + 1];
      if (tm > -INFINITY) sum += tsv * __expf(tm - m);
    }
#pragma unroll
    for (int d = 1; d < 32; d <<= 1) sum += __shfl_xor(sum, d, 32);
    lm[s] = m; ls[s] = sum;
  }

  const int tgt = target[row];
  const float* h  = hidden + (size_t)row * DK;
  const float* wt = weight + (size_t)tgt * DK;
  float dt = 0.0f, d0 = 0.0f, d1 = 0.0f, d2 = 0.0f;
  for (int k = lane; k < DK; k += 32) {
    float hv = h[k];
    dt += hv * wt[k];
    d0 += hv * cw[k];
    d1 += hv * cw[DK + k];
    d2 += hv * cw[2 * DK + k];
  }
#pragma unroll
  for (int d = 1; d < 32; d <<= 1) {
    dt += __shfl_xor(dt, d, 32);
    d0 += __shfl_xor(d0, d, 32);
    d1 += __shfl_xor(d1, d, 32);
    d2 += __shfl_xor(d2, d, 32);
  }

  if (lane == 0) {
    float r0 = d0 + cbias[0], r1 = d1 + cbias[1], r2 = d2 + cbias[2];
    float M = fmaxf(fmaxf(lm[0], r0), fmaxf(r1, r2));
    float S = __expf(r0 - M) + __expf(r1 - M) + __expf(r2 - M);
    if (lm[0] > -INFINITY) S += ls[0] * __expf(lm[0] - M);
    float lseH = M + __logf(S);
    float tl = dt + bias[tgt];
    float nll;
    if (tgt < 5000) {
      nll = lseH - tl;
    } else {
      int s = (tgt < 15000) ? 1 : ((tgt < 30000) ? 2 : 3);
      float lmS = (s == 1) ? lm[1] : ((s == 2) ? lm[2] : lm[3]);
      float lsS = (s == 1) ? ls[1] : ((s == 2) ? ls[2] : ls[3]);
      float lseT = lmS + __logf(lsS);
      // reference: cluster i routes via head column -i  ->  r[3 - i]
      float route = ((s == 1) ? r2 : ((s == 2) ? r1 : r0)) - lseH;
      nll = -(route + (tl - lseT));
    }
    out[row] = nll;
  }
}

extern "C" void kernel_launch(void* const* d_in, const int* in_sizes, int n_in,
                              void* d_out, int out_size, void* d_ws, size_t ws_size,
                              hipStream_t stream) {
  const float* hidden = (const float*)d_in[0];
  const int*   target = (const int*)d_in[1];
  const float* weight = (const float*)d_in[2];
  const float* bias   = (const float*)d_in[3];
  const float* cw     = (const float*)d_in[4];
  const float* cbias  = (const float*)d_in[5];
  float* out = (float*)d_out;

  // workspace layout: partials | bf16 hidden | bf16 weight  (~62 MB total)
  float* part = (float*)d_ws;
  unsigned short* hb = (unsigned short*)((char*)d_ws +
                        (size_t)NTILES * NROWS * 2 * sizeof(float));
  unsigned short* wb = hb + (size_t)NROWS * DK;

  const int nh4 = NROWS * DK / 4;            // 524288
  const int nw4 = 50257 * DK / 4;            // 6432896
  cvt_bf16_kernel<<<(nh4 + 255) / 256, 256, 0, stream>>>(hidden, hb, nh4);
  cvt_bf16_kernel<<<(nw4 + 255) / 256, 256, 0, stream>>>(weight, wb, nw4);

  dim3 g1(NTILES, NROWS / MT);
  adasoft_gemm_lse<<<g1, 256, 0, stream>>>(hb, wb, bias, part);
  adasoft_finalize<<<NROWS / 8, 256, 0, stream>>>(hidden, target, weight, bias,
                                                  cw, cbias, part, out);
}
